// BahdanauTemporalAttention_2156073583149
// MI455X (gfx1250) — matrix operand order repaired and checked by reading, not compiled, not run on hardware
//
#include <hip/hip_runtime.h>

// Problem dims (fixed by the reference)
#define B_  32
#define S_  4096
#define H_  1024
#define A_  512
#define KC  64          // K-chunk staged in LDS per iteration
#define PITCH 72        // LDS row pitch in bf16 (144 B: 16-B aligned, conflict-free)

typedef __attribute__((ext_vector_type(16))) __bf16 v16bf;
typedef __attribute__((ext_vector_type(2)))  __bf16 v2bf;
typedef __attribute__((ext_vector_type(8)))  float  v8f;
typedef __attribute__((ext_vector_type(4)))  int    intv4;

__device__ __forceinline__ __bf16 f2bf(float x) { return (__bf16)x; }

// packs to a single v_cvt_pk_bf16_f32
__device__ __forceinline__ unsigned pk_bf16(float x, float y) {
  v2bf v = { (__bf16)x, (__bf16)y };
  return __builtin_bit_cast(unsigned, v);
}

// tanh via exp: a handful of VALU ops instead of the libm branch tree.
__device__ __forceinline__ float fast_tanh(float x) {
  float xc = fminf(fmaxf(x, -9.0f), 9.0f);
  float e  = __expf(2.0f * xc);
  return (e - 1.0f) / (e + 1.0f);
}

struct Frag32 { uint4 a, b; };   // 32 bytes == one v16bf fragment

// CDNA5 async global->LDS copy (guarded; falls back to a plain copy)
#if __has_builtin(__builtin_amdgcn_global_load_async_to_lds_b128) && \
    __has_builtin(__builtin_amdgcn_s_wait_asynccnt)
#define USE_ASYNC_LDS 1
#endif

__device__ __forceinline__ void copy16_g2lds(void* lds, const void* g) {
#ifdef USE_ASYNC_LDS
  __builtin_amdgcn_global_load_async_to_lds_b128(
      (__attribute__((address_space(1))) intv4*)(g),
      (__attribute__((address_space(3))) intv4*)(lds), 0, 0);
#else
  *(uint4*)lds = *(const uint4*)g;
#endif
}

__device__ __forceinline__ void wait_async_copies() {
#ifdef USE_ASYNC_LDS
  __builtin_amdgcn_s_wait_asynccnt(0);
#endif
}

// ---------------------------------------------------------------------------
// Kernel 1: WkT[a][h] = bf16(Wk[h][a])   (one-time 32x32-tiled transpose)
// ---------------------------------------------------------------------------
__global__ void wk_transpose_kernel(const float* __restrict__ wk,
                                    __bf16* __restrict__ wkT) {
  __shared__ float tile[32][33];
  const int hb = blockIdx.x * 32, ab = blockIdx.y * 32;
  const int tx = threadIdx.x & 31, ty = threadIdx.x >> 5;   // ty: 0..7
  #pragma unroll
  for (int i = ty; i < 32; i += 8)
    tile[i][tx] = wk[(size_t)(hb + i) * A_ + ab + tx];
  __syncthreads();
  #pragma unroll
  for (int i = ty; i < 32; i += 8)
    wkT[(size_t)(ab + i) * H_ + hb + tx] = f2bf(tile[tx][i]);
}

// ---------------------------------------------------------------------------
// Kernel 2: q_proj[b][a] = sum_h query[b][h] * Wq[h][a]   (32 x 512, tiny)
// ---------------------------------------------------------------------------
__global__ void qproj_kernel(const float* __restrict__ q,
                             const float* __restrict__ wq,
                             float* __restrict__ qv) {
  const int b = blockIdx.x;
  const int a = blockIdx.y * 256 + threadIdx.x;
  float acc = 0.f;
  #pragma unroll 4
  for (int h = 0; h < H_; ++h)
    acc = fmaf(q[b * H_ + h], wq[(size_t)h * A_ + a], acc);
  qv[b * A_ + a] = acc;
}

// ---------------------------------------------------------------------------
// Kernel 3: fused  energy[b][s] = sum_a tanh(qv[b][a] + keys[b][s]@Wk[:,a]) v[a]
// One workgroup (8 wave32s) per (batch, 32-row S tile).
// M=32, N=512 (all of A in VGPR accumulators), K staged 64 at a time in LDS.
// All 8 B fragments of a K-substep are preloaded into distinct registers so
// the 16 ds_load_b128 issue back-to-back and overlap the 8 WMMAs.
// ---------------------------------------------------------------------------
__launch_bounds__(256, 2)
__global__ void energy_kernel(const float* __restrict__ keys,
                              const __bf16* __restrict__ wkT,
                              const float* __restrict__ qv,
                              const float* __restrict__ v,
                              float* __restrict__ energy) {
  __shared__ __align__(16) __bf16 lds_keys[32][PITCH];    // keys tile (K-major)
  __shared__ __align__(16) __bf16 lds_wkT[512][PITCH];    // WkT tile [N][K], 72 KB
  __shared__ float lds_energy[32];

  const int tid  = threadIdx.x;
  const int wave = tid >> 5;
  const int lane = tid & 31;

  const int b      = blockIdx.x >> 7;            // S_/32 == 128 tiles per batch
  const int s_base = (blockIdx.x & 127) * 32;

  if (tid < 32) lds_energy[tid] = 0.f;

  const int m    = wave & 1;      // 16-row half of the 32-row S tile
  const int nb   = wave >> 1;     // 0..3 interleaved N-tile ownership
  const int lrow = lane & 15;
  const int kgrp = lane >> 4;     // half-wave (K-split per ISA 7.12.2)

  v8f acc[8];
  #pragma unroll
  for (int j = 0; j < 8; ++j) acc[j] = (v8f){0.f,0.f,0.f,0.f,0.f,0.f,0.f,0.f};

  const float* keys_b = keys + ((size_t)b * S_ + s_base) * H_;

  for (int k0 = 0; k0 < H_; k0 += KC) {
    __syncthreads();
    // --- stage keys 32x64 fp32 -> bf16 (two float4 loads, one b128 store) ---
    {
      const int r  = tid >> 3;            // 0..31
      const int c0 = (tid & 7) * 8;       // 0,8,...,56
      const float* src = keys_b + (size_t)r * H_ + k0 + c0;
      const float4 f0 = *(const float4*)(src);
      const float4 f1 = *(const float4*)(src + 4);
      uint4 p;
      p.x = pk_bf16(f0.x, f0.y);
      p.y = pk_bf16(f0.z, f0.w);
      p.z = pk_bf16(f1.x, f1.y);
      p.w = pk_bf16(f1.z, f1.w);
      *(uint4*)&lds_keys[r][c0] = p;
    }
    // --- stage WkT tile: 512 rows x 64 K (128 B/row) via async b128 ---------
    #pragma unroll
    for (int i = 0; i < 16; ++i) {
      const int idx = tid + i * 256;        // 0..4095 16-B chunks
      const int row = idx >> 3;             // 0..511  (N)
      const int c   = (idx & 7) * 8;        // bf16 offset 0..56
      copy16_g2lds(&lds_wkT[row][c], wkT + (size_t)row * H_ + k0 + c);
    }
    wait_async_copies();
    __syncthreads();

    // --- two K=32 sub-steps out of the staged 64 ----------------------------
    #pragma unroll
    for (int ks = 0; ks < 2; ++ks) {
      const int kk = ks * 32;
      v16bf afrag;
      {
        const __bf16* arow = &lds_keys[m * 16 + lrow][kk];
        Frag32 fa;
        fa.a = *(const uint4*)(arow + 16 * kgrp);       // K = 16*kgrp..+7
        fa.b = *(const uint4*)(arow + 16 * kgrp + 8);   // K = 16*kgrp+8..+15
        afrag = __builtin_bit_cast(v16bf, fa);
      }
      // preload ALL 8 B fragments into distinct registers (16 ds_load_b128
      // issued back-to-back -> staged s_wait_dscnt, latency overlapped)
      Frag32 fb[8];
      #pragma unroll
      for (int j = 0; j < 8; ++j) {
        const __bf16* p = &lds_wkT[(nb + 4 * j) * 16 + lrow][kk + 16 * kgrp];
        fb[j].a = ((const uint4*)p)[0];
        fb[j].b = ((const uint4*)p)[1];
      }
      #pragma unroll
      for (int j = 0; j < 8; ++j) {
        acc[j] = __builtin_amdgcn_wmma_f32_16x16x32_bf16(
            false, afrag, false, __builtin_bit_cast(v16bf, fb[j]),
            (short)0, acc[j], false, false);
      }
    }
  }
  __syncthreads();

  // --- epilogue: tanh(acc + q_proj) * v, reduce over A ----------------------
  const float* qv_b = qv + b * A_;
  float qa[8], vv[8];
  #pragma unroll
  for (int j = 0; j < 8; ++j) {
    const int acol = (nb + 4 * j) * 16 + lrow;
    qa[j] = qv_b[acol];
    vv[j] = v[acol];
  }
  #pragma unroll
  for (int i = 0; i < 8; ++i) {
    // per-lane partial over this lane's 8 A-columns first (s fixed per i)
    float partial = 0.f;
    #pragma unroll
    for (int j = 0; j < 8; ++j)
      partial += fast_tanh(acc[j][i] + qa[j]) * vv[j];
    // one cross-lane reduction per i (16 lanes share the same s-row)
    partial += __shfl_xor(partial, 1, 32);
    partial += __shfl_xor(partial, 2, 32);
    partial += __shfl_xor(partial, 4, 32);
    partial += __shfl_xor(partial, 8, 32);
    if (lrow == 0) {
      const int s_local = m * 16 + i + 8 * kgrp;
      atomicAdd(&lds_energy[s_local], partial);       // ds_add_f32
    }
  }
  __syncthreads();
  if (tid < 32)
    energy[(size_t)b * S_ + s_base + tid] = lds_energy[tid];
}

// ---------------------------------------------------------------------------
// Kernel 4: in-place softmax over S per batch row
// ---------------------------------------------------------------------------
__global__ void softmax_kernel(float* __restrict__ e) {
  __shared__ float red[256];
  const int b = blockIdx.x;
  const int t = threadIdx.x;
  float* row = e + (size_t)b * S_;

  float mx = -3.402823466e38f;
  for (int s = t; s < S_; s += 256) mx = fmaxf(mx, row[s]);
  red[t] = mx; __syncthreads();
  for (int off = 128; off > 0; off >>= 1) {
    if (t < off) red[t] = fmaxf(red[t], red[t + off]);
    __syncthreads();
  }
  mx = red[0]; __syncthreads();

  float sum = 0.f;
  for (int s = t; s < S_; s += 256) sum += __expf(row[s] - mx);
  red[t] = sum; __syncthreads();
  for (int off = 128; off > 0; off >>= 1) {
    if (t < off) red[t] += red[t + off];
    __syncthreads();
  }
  const float inv = 1.0f / red[0];
  __syncthreads();
  for (int s = t; s < S_; s += 256) row[s] = __expf(row[s] - mx) * inv;
}

// ---------------------------------------------------------------------------
// Kernel 5a: partial context over S chunks (deterministic two-stage reduce)
// ---------------------------------------------------------------------------
__global__ void context_partial_kernel(const float* __restrict__ attn,
                                       const float* __restrict__ keys,
                                       float* __restrict__ part) {
  const int b  = blockIdx.x;
  const int h  = blockIdx.y * 256 + threadIdx.x;
  const int sc = blockIdx.z;
  const int s0 = sc * (S_ / 8);
  const float* arow = attn + (size_t)b * S_;
  const float* kcol = keys + (size_t)b * S_ * H_ + h;
  float acc = 0.f;
  #pragma unroll 4
  for (int s = s0; s < s0 + S_ / 8; ++s)
    acc = fmaf(arow[s], kcol[(size_t)s * H_], acc);
  part[((size_t)b * 8 + sc) * H_ + h] = acc;
}

// Kernel 5b: reduce the 8 partials -> out (B_, H_)
__global__ void context_reduce_kernel(const float* __restrict__ part,
                                      float* __restrict__ out) {
  const int i = blockIdx.x * 256 + threadIdx.x;   // 0 .. B_*H_-1
  const int b = i / H_, h = i - b * H_;
  float acc = 0.f;
  #pragma unroll
  for (int sc = 0; sc < 8; ++sc)
    acc += part[((size_t)b * 8 + sc) * H_ + h];
  out[i] = acc;
}

// ---------------------------------------------------------------------------
// Launch
// ---------------------------------------------------------------------------
extern "C" void kernel_launch(void* const* d_in, const int* in_sizes, int n_in,
                              void* d_out, int out_size, void* d_ws, size_t ws_size,
                              hipStream_t stream) {
  (void)in_sizes; (void)n_in; (void)out_size; (void)ws_size;

  const float* query = (const float*)d_in[0];   // (B, H)
  const float* keys  = (const float*)d_in[1];   // (B, S, H)
  const float* Wq    = (const float*)d_in[2];   // (H, A)
  const float* Wk    = (const float*)d_in[3];   // (H, A)
  const float* v     = (const float*)d_in[4];   // (A,)

  // workspace layout
  char*   ws     = (char*)d_ws;
  float*  qv     = (float*)ws;                                   //  64 KB
  __bf16* wkT    = (__bf16*)(ws + (size_t)B_ * A_ * 4);          //   1 MB
  float*  energy = (float*)(ws + (size_t)B_ * A_ * 4
                               + (size_t)H_ * A_ * 2);           // 512 KB
  float*  part   = (float*)(ws + (size_t)B_ * A_ * 4
                               + (size_t)H_ * A_ * 2
                               + (size_t)B_ * S_ * 4);           //   1 MB

  wk_transpose_kernel<<<dim3(H_ / 32, A_ / 32), 256, 0, stream>>>(Wk, wkT);
  qproj_kernel<<<dim3(B_, A_ / 256), 256, 0, stream>>>(query, Wq, qv);
  energy_kernel<<<B_ * (S_ / 32), 256, 0, stream>>>(keys, wkT, qv, v, energy);
  softmax_kernel<<<B_, 256, 0, stream>>>(energy);
  context_partial_kernel<<<dim3(B_, H_ / 256, 8), 256, 0, stream>>>(energy, keys, part);
  context_reduce_kernel<<<(B_ * H_) / 256, 256, 0, stream>>>(part, d_out ? (float*)d_out : part);
}
